// GraphConvResBlock_11184094839562
// MI455X (gfx1250) — matrix operand order repaired and checked by reading, not compiled, not run on hardware
//
#include <hip/hip_runtime.h>
#include <hip/hip_bf16.h>

// ---------------------------------------------------------------------------
// GraphConvResBlock, restructured (adj@(X@W) == (adj@X)@W):
//   ah  = adj @ BN1(x)                       (bf16 WMMA, f32 accum)
//   skip = ah @ w_skip + b_skip              -> d_out (f32)
//   t1b = BN2(lrelu(ah @ w1 + b1))           (bf16)
//   a2  = adj @ t1b                          (bf16 WMMA)
//   d_out += lrelu(a2 @ w2 + b2)
// GEMMs use V_WMMA_F32_16X16X32_BF16 with double-buffered LDS staged by
// GLOBAL_LOAD_ASYNC_TO_LDS_B128 (ASYNCcnt), the CDNA5 async-copy path.
// Workspace use: ~67.5 MB.
// ---------------------------------------------------------------------------

typedef __bf16 bf16;
typedef __attribute__((ext_vector_type(4)))  __bf16 v4bf;
typedef __attribute__((ext_vector_type(8)))  __bf16 v8bf;
typedef __attribute__((ext_vector_type(16))) __bf16 v16bf;
typedef __attribute__((ext_vector_type(8)))  float  v8f;

#define NB 16     // batch
#define NN 1024   // nodes
#define NC 256    // channels

__device__ __forceinline__ bf16 f2bf(float f) {
  union { float f; unsigned u; } c; c.f = f;
  unsigned r = c.u + 0x7FFFu + ((c.u >> 16) & 1u);   // round-to-nearest-even
  unsigned short h = (unsigned short)(r >> 16);
  union { unsigned short s; bf16 b; } o; o.s = h;
  return o.b;
}

__device__ __forceinline__ v8f vzero8() {
  v8f z;
#pragma unroll
  for (int i = 0; i < 8; ++i) z[i] = 0.0f;
  return z;
}

// CDNA5 async copy: global -> LDS, 16B per lane, tracked by ASYNCcnt.
__device__ __forceinline__ void async_ld_b128(unsigned lds_addr, const bf16* gptr) {
  asm volatile("global_load_async_to_lds_b128 %0, %1, off"
               :: "v"(lds_addr), "v"(gptr) : "memory");
}
__device__ __forceinline__ void wait_async0() {
  asm volatile("s_wait_asynccnt 0x0" ::: "memory");
}
__device__ __forceinline__ unsigned lds_off(const bf16* p) {
  return (unsigned)(size_t)p;   // low 32 bits of generic ptr = LDS byte address
}

// A fragment: 16x32 bf16 tile, row-major in LDS with leading dim ld (elements).
// lane%16 = M row; elems 0-7 = K[(lane/16)*8,+8); elems 8-15 = K[16+(lane/16)*8,+8)
__device__ __forceinline__ v16bf load_frag_a(const bf16* base, int lane, int ld) {
  const int r  = lane & 15;
  const int k0 = (lane >> 4) << 3;
  v8bf lo = *(const v8bf*)(base + r * ld + k0);
  v8bf hi = *(const v8bf*)(base + r * ld + 16 + k0);
  return __builtin_shufflevector(lo, hi, 0,1,2,3,4,5,6,7,8,9,10,11,12,13,14,15);
}

// B fragment: 32x16 bf16 tile, row-major; lane%16 = N column (nb + lane%16),
// elems 0-7 = K[(lane/16)*8,+8); elems 8-15 = K[16+(lane/16)*8,+8)
__device__ __forceinline__ v16bf load_frag_b(const bf16* base, int lane, int nb, int ld) {
  const bf16* p = base + nb + (lane & 15);
  const int k0 = (lane >> 4) << 3;
  v16bf v;
#pragma unroll
  for (int i = 0; i < 8; ++i) {
    v[i]     = p[(k0 + i) * ld];
    v[8 + i] = p[(16 + k0 + i) * ld];
  }
  return v;
}

__device__ __forceinline__ v8f wmma_bf(v16bf a, v16bf b, v8f c) {
  return __builtin_amdgcn_wmma_f32_16x16x32_bf16(false, a, false, b, (short)0, c,
                                                 false, false);
}

// ------------------------- elementwise kernels -----------------------------

__global__ void k_cast_bf16(const float* __restrict__ src, bf16* __restrict__ dst,
                            int n4) {
  int i = blockIdx.x * blockDim.x + threadIdx.x;
  if (i >= n4) return;
  float4 v = ((const float4*)src)[i];
  v4bf o;
  o[0] = f2bf(v.x); o[1] = f2bf(v.y); o[2] = f2bf(v.z); o[3] = f2bf(v.w);
  ((v4bf*)dst)[i] = o;
}

__global__ void k_bn1_cast(const float* __restrict__ x,
                           const float* __restrict__ g, const float* __restrict__ be,
                           const float* __restrict__ mu, const float* __restrict__ var,
                           bf16* __restrict__ h, int n4) {
  int i = blockIdx.x * blockDim.x + threadIdx.x;
  if (i >= n4) return;
  int c0 = (i * 4) & (NC - 1);
  float4 v = ((const float4*)x)[i];
  float vals[4] = {v.x, v.y, v.z, v.w};
  v4bf o;
#pragma unroll
  for (int j = 0; j < 4; ++j) {
    int c = c0 + j;
    float s = g[c] * rsqrtf(var[c] + 1e-3f);
    o[j] = f2bf((vals[j] - mu[c]) * s + be[c]);
  }
  ((v4bf*)h)[i] = o;
}

// ------------------- adjacency GEMM: O = A(1024x1024) @ B(1024x256) --------
// 128x128 block tile, BK=64, 8 waves (2x4), wave tile 64x32 (4x2 WMMA tiles).
// Double-buffered LDS filled by async loads; 16 K-stages.

#define ALDA 72    // 64 + 8 pad (elements)
#define ALDB 136   // 128 + 8 pad

__device__ __forceinline__ void adj_stage(unsigned aB, unsigned bB,
                                          const bf16* gA, const bf16* gB,
                                          int tid, int mBase, int nBase, int kb) {
#pragma unroll
  for (int i = 0; i < 4; ++i) {                   // A tile: 128 x 64
    int idx = tid + i * 256;                      // 0..1023
    int r = idx >> 3, c = (idx & 7) << 3;
    async_ld_b128(aB + (unsigned)(r * ALDA + c) * 2,
                  gA + (size_t)(mBase + r) * NN + kb + c);
  }
#pragma unroll
  for (int i = 0; i < 4; ++i) {                   // B tile: 64 x 128
    int idx = tid + i * 256;
    int r = idx >> 4, c = (idx & 15) << 3;
    async_ld_b128(bB + (unsigned)(r * ALDB + c) * 2,
                  gB + (size_t)(kb + r) * NC + nBase + c);
  }
}

__device__ __forceinline__ void adj_compute(const bf16* sAp, const bf16* sBp,
                                            int lane, int wm, int wn,
                                            v8f (&acc)[4][2]) {
#pragma unroll
  for (int ks = 0; ks < 64; ks += 32) {
    v16bf bfr[2];
#pragma unroll
    for (int n = 0; n < 2; ++n)
      bfr[n] = load_frag_b(sBp + ks * ALDB, lane, wn * 32 + n * 16, ALDB);
#pragma unroll
    for (int m = 0; m < 4; ++m) {
      v16bf af = load_frag_a(sAp + (wm * 64 + m * 16) * ALDA + ks, lane, ALDA);
#pragma unroll
      for (int n = 0; n < 2; ++n) acc[m][n] = wmma_bf(af, bfr[n], acc[m][n]);
    }
  }
}

__global__ __launch_bounds__(256) void k_gemm_adj(const bf16* __restrict__ A,
                                                  const bf16* __restrict__ Bm,
                                                  bf16* __restrict__ O) {
  __shared__ __align__(16) bf16 sA[2][128 * ALDA];
  __shared__ __align__(16) bf16 sB[2][64 * ALDB];

  const int b     = blockIdx.z;
  const int mBase = blockIdx.x * 128;
  const int nBase = blockIdx.y * 128;
  const bf16* gA = A  + (size_t)b * NN * NN;
  const bf16* gB = Bm + (size_t)b * NN * NC;
  bf16*       gO = O  + (size_t)b * NN * NC;

  const int tid  = threadIdx.x;
  const int lane = tid & 31;
  const int wave = tid >> 5;      // 0..7
  const int wm   = wave >> 2;     // 0..1 : 64 rows
  const int wn   = wave & 3;      // 0..3 : 32 cols

  const unsigned a0 = lds_off(&sA[0][0]), a1 = lds_off(&sA[1][0]);
  const unsigned b0 = lds_off(&sB[0][0]), b1 = lds_off(&sB[1][0]);

  v8f acc[4][2];
#pragma unroll
  for (int m = 0; m < 4; ++m)
#pragma unroll
    for (int n = 0; n < 2; ++n) acc[m][n] = vzero8();

  adj_stage(a0, b0, gA, gB, tid, mBase, nBase, 0);
  wait_async0();
  __syncthreads();

  for (int kt = 0; kt < 16; kt += 2) {            // 16 stages of BK=64
    adj_stage(a1, b1, gA, gB, tid, mBase, nBase, (kt + 1) * 64);
    adj_compute(&sA[0][0], &sB[0][0], lane, wm, wn, acc);
    wait_async0();
    __syncthreads();

    if (kt + 2 < 16)
      adj_stage(a0, b0, gA, gB, tid, mBase, nBase, (kt + 2) * 64);
    adj_compute(&sA[1][0], &sB[1][0], lane, wm, wn, acc);
    wait_async0();
    __syncthreads();
  }

  const int rofs = (lane >> 4) << 3;
  const int cofs = lane & 15;
#pragma unroll
  for (int m = 0; m < 4; ++m)
#pragma unroll
    for (int n = 0; n < 2; ++n) {
      int col = nBase + wn * 32 + n * 16 + cofs;
#pragma unroll
      for (int v = 0; v < 8; ++v) {
        int row = mBase + wm * 64 + m * 16 + rofs + v;
        gO[(size_t)row * NC + col] = f2bf(acc[m][n][v]);
      }
    }
}

// ------ dual feature GEMM: ah@w1 (->BN2(lrelu)) and ah@w_skip (->d_out) ----
// 128x64 block tile, BK=32, 8 waves (2x4), wave tile 64x16, shared A frags,
// async double-buffered staging; 8 K-stages.

#define FLDA 40    // 32 + 8 pad
#define FLDB 72    // 64 + 8 pad

__device__ __forceinline__ void feat_stage_a(unsigned aB, const bf16* gA,
                                             int tid, int mBase, int kb) {
#pragma unroll
  for (int i = 0; i < 2; ++i) {                   // A tile: 128 x 32
    int idx = tid + i * 256;
    int r = idx >> 2, c = (idx & 3) << 3;
    async_ld_b128(aB + (unsigned)(r * FLDA + c) * 2,
                  gA + (size_t)(mBase + r) * NC + kb + c);
  }
}

__device__ __forceinline__ void feat_stage_b(unsigned bB, const bf16* W,
                                             int tid, int nBase, int kb) {
  int r = tid >> 3, c = (tid & 7) << 3;           // B tile: 32 x 64
  async_ld_b128(bB + (unsigned)(r * FLDB + c) * 2,
                W + (size_t)(kb + r) * NC + nBase + c);
}

__global__ __launch_bounds__(256) void k_gemm_feat_dual(
    const bf16* __restrict__ A, const bf16* __restrict__ W1,
    const bf16* __restrict__ WS, const float* __restrict__ b1,
    const float* __restrict__ g2, const float* __restrict__ be2,
    const float* __restrict__ mu2, const float* __restrict__ var2,
    const float* __restrict__ bS, bf16* __restrict__ T1B,
    float* __restrict__ OUT) {
  __shared__ __align__(16) bf16 sA[2][128 * FLDA];
  __shared__ __align__(16) bf16 sB1[2][32 * FLDB];
  __shared__ __align__(16) bf16 sBS[2][32 * FLDB];

  const int b     = blockIdx.z;
  const int mBase = blockIdx.x * 128;
  const int nBase = blockIdx.y * 64;
  const bf16* gA = A + (size_t)b * NN * NC;
  bf16*  gT = T1B + (size_t)b * NN * NC;
  float* gO = OUT + (size_t)b * NN * NC;

  const int tid  = threadIdx.x;
  const int lane = tid & 31;
  const int wave = tid >> 5;
  const int wm   = wave >> 2;    // 0..1
  const int wn   = wave & 3;     // 0..3 : 16 cols each

  const unsigned a0 = lds_off(&sA[0][0]),  a1 = lds_off(&sA[1][0]);
  const unsigned p0 = lds_off(&sB1[0][0]), p1 = lds_off(&sB1[1][0]);
  const unsigned q0 = lds_off(&sBS[0][0]), q1 = lds_off(&sBS[1][0]);

  v8f acc1[4], accS[4];
#pragma unroll
  for (int m = 0; m < 4; ++m) { acc1[m] = vzero8(); accS[m] = vzero8(); }

  feat_stage_a(a0, gA, tid, mBase, 0);
  feat_stage_b(p0, W1, tid, nBase, 0);
  feat_stage_b(q0, WS, tid, nBase, 0);
  wait_async0();
  __syncthreads();

  for (int kt = 0; kt < 8; kt += 2) {             // 8 stages of BK=32
#pragma unroll
    for (int half = 0; half < 2; ++half) {
      const int nxt = kt + half + 1;
      if (nxt < 8) {
        feat_stage_a(half ? a0 : a1, gA, tid, mBase, nxt * 32);
        feat_stage_b(half ? p0 : p1, W1, tid, nBase, nxt * 32);
        feat_stage_b(half ? q0 : q1, WS, tid, nBase, nxt * 32);
      }
      const bf16* sAp  = &sA[half][0];
      v16bf b1f = load_frag_b(&sB1[half][0], lane, wn * 16, FLDB);
      v16bf bSf = load_frag_b(&sBS[half][0], lane, wn * 16, FLDB);
#pragma unroll
      for (int m = 0; m < 4; ++m) {
        v16bf af = load_frag_a(sAp + (wm * 64 + m * 16) * FLDA, lane, FLDA);
        acc1[m] = wmma_bf(af, b1f, acc1[m]);
        accS[m] = wmma_bf(af, bSf, accS[m]);
      }
      wait_async0();
      __syncthreads();
    }
  }

  const int col   = nBase + wn * 16 + (lane & 15);
  const float bia = b1[col];
  const float scl = g2[col] * rsqrtf(var2[col] + 1e-3f);
  const float sft = be2[col] - mu2[col] * scl;
  const float bsk = bS[col];
  const int rofs  = (lane >> 4) << 3;
#pragma unroll
  for (int m = 0; m < 4; ++m)
#pragma unroll
    for (int v = 0; v < 8; ++v) {
      int row = mBase + wm * 64 + m * 16 + rofs + v;
      float t = acc1[m][v] + bia;
      t = (t >= 0.0f) ? t : 0.3f * t;             // LeakyReLU(0.3)
      gT[(size_t)row * NC + col] = f2bf(t * scl + sft);   // BN2 fused
      gO[(size_t)row * NC + col] = accS[m][v] + bsk;      // skip path (f32)
    }
}

// ----------- final feature GEMM: d_out += lrelu(a2 @ w2 + b2) --------------

__global__ __launch_bounds__(256) void k_gemm_feat_final(
    const bf16* __restrict__ A, const bf16* __restrict__ W2,
    const float* __restrict__ b2, float* __restrict__ OUT) {
  __shared__ __align__(16) bf16 sA[2][128 * FLDA];
  __shared__ __align__(16) bf16 sB[2][32 * FLDB];

  const int b     = blockIdx.z;
  const int mBase = blockIdx.x * 128;
  const int nBase = blockIdx.y * 64;
  const bf16* gA = A + (size_t)b * NN * NC;
  float* gO = OUT + (size_t)b * NN * NC;

  const int tid  = threadIdx.x;
  const int lane = tid & 31;
  const int wave = tid >> 5;
  const int wm   = wave >> 2;
  const int wn   = wave & 3;

  const unsigned a0 = lds_off(&sA[0][0]), a1 = lds_off(&sA[1][0]);
  const unsigned p0 = lds_off(&sB[0][0]), p1 = lds_off(&sB[1][0]);

  v8f acc[4];
#pragma unroll
  for (int m = 0; m < 4; ++m) acc[m] = vzero8();

  feat_stage_a(a0, gA, tid, mBase, 0);
  feat_stage_b(p0, W2, tid, nBase, 0);
  wait_async0();
  __syncthreads();

  for (int kt = 0; kt < 8; kt += 2) {
#pragma unroll
    for (int half = 0; half < 2; ++half) {
      const int nxt = kt + half + 1;
      if (nxt < 8) {
        feat_stage_a(half ? a0 : a1, gA, tid, mBase, nxt * 32);
        feat_stage_b(half ? p0 : p1, W2, tid, nBase, nxt * 32);
      }
      const bf16* sAp = &sA[half][0];
      v16bf bf = load_frag_b(&sB[half][0], lane, wn * 16, FLDB);
#pragma unroll
      for (int m = 0; m < 4; ++m) {
        v16bf af = load_frag_a(sAp + (wm * 64 + m * 16) * FLDA, lane, FLDA);
        acc[m] = wmma_bf(af, bf, acc[m]);
      }
      wait_async0();
      __syncthreads();
    }
  }

  const int col  = nBase + wn * 16 + (lane & 15);
  const float bi = b2[col];
  const int rofs = (lane >> 4) << 3;
#pragma unroll
  for (int m = 0; m < 4; ++m)
#pragma unroll
    for (int v = 0; v < 8; ++v) {
      int row = mBase + wm * 64 + m * 16 + rofs + v;
      float t = acc[m][v] + bi;
      t = (t >= 0.0f) ? t : 0.3f * t;
      size_t idx = (size_t)row * NC + col;
      gO[idx] = gO[idx] + t;                       // += out2 after skip stored
    }
}

// ---------------------------------------------------------------------------

extern "C" void kernel_launch(void* const* d_in, const int* in_sizes, int n_in,
                              void* d_out, int out_size, void* d_ws, size_t ws_size,
                              hipStream_t stream) {
  const float* x    = (const float*)d_in[0];
  const float* adj  = (const float*)d_in[1];
  const float* g1   = (const float*)d_in[2];
  const float* be1  = (const float*)d_in[3];
  const float* mu1  = (const float*)d_in[4];
  const float* var1 = (const float*)d_in[5];
  const float* w1   = (const float*)d_in[6];
  const float* b1   = (const float*)d_in[7];
  const float* g2   = (const float*)d_in[8];
  const float* be2  = (const float*)d_in[9];
  const float* mu2  = (const float*)d_in[10];
  const float* var2 = (const float*)d_in[11];
  const float* w2   = (const float*)d_in[12];
  const float* b2   = (const float*)d_in[13];
  const float* wS   = (const float*)d_in[14];
  const float* bS   = (const float*)d_in[15];
  float* out = (float*)d_out;

  char* ws = (char*)d_ws;
  size_t off = 0;
  auto alloc = [&](size_t bytes) {
    char* p = ws + off;
    off += (bytes + 255) & ~(size_t)255;
    return p;
  };
  bf16* adjb = (bf16*)alloc((size_t)NB * NN * NN * 2);   // 33.5 MB
  bf16* hb   = (bf16*)alloc((size_t)NB * NN * NC * 2);   // 8.4 MB
  bf16* ahb  = (bf16*)alloc((size_t)NB * NN * NC * 2);
  bf16* t1b  = (bf16*)alloc((size_t)NB * NN * NC * 2);
  bf16* a2b  = (bf16*)alloc((size_t)NB * NN * NC * 2);
  bf16* w1b  = (bf16*)alloc((size_t)NC * NC * 2);
  bf16* w2b  = (bf16*)alloc((size_t)NC * NC * 2);
  bf16* wSb  = (bf16*)alloc((size_t)NC * NC * 2);

  {
    int n4 = NB * NN * NN / 4;
    k_cast_bf16<<<(n4 + 255) / 256, 256, 0, stream>>>(adj, adjb, n4);
  }
  {
    int n4 = NC * NC / 4;
    k_cast_bf16<<<(n4 + 255) / 256, 256, 0, stream>>>(w1, w1b, n4);
    k_cast_bf16<<<(n4 + 255) / 256, 256, 0, stream>>>(w2, w2b, n4);
    k_cast_bf16<<<(n4 + 255) / 256, 256, 0, stream>>>(wS, wSb, n4);
  }
  {
    int n4 = NB * NN * NC / 4;
    k_bn1_cast<<<(n4 + 255) / 256, 256, 0, stream>>>(x, g1, be1, mu1, var1, hb, n4);
  }

  dim3 gAdj(NN / 128, NC / 128, NB);   // (8, 2, 16)
  dim3 gFeat(NN / 128, NC / 64, NB);   // (8, 4, 16)

  k_gemm_adj<<<gAdj, 256, 0, stream>>>(adjb, hb, ahb);               // ah = adj@h
  k_gemm_feat_dual<<<gFeat, 256, 0, stream>>>(ahb, w1b, wSb, b1, g2, be2,
                                              mu2, var2, bS, t1b, out);
  k_gemm_adj<<<gAdj, 256, 0, stream>>>(adjb, t1b, a2b);              // a2 = adj@t1b
  k_gemm_feat_final<<<gFeat, 256, 0, stream>>>(a2b, w2b, b2, out);
}
